// CausalWanSelfAttention_7902739825182
// MI455X (gfx1250) — compile-verified, matrix operand order checked by reading, not run on hardware
//
#include <hip/hip_runtime.h>

#define DIMN 1536
#define NHN  12
#define HDN  128
#define S_LEN 4680
#define SP   4736            // padded rows: 37*128
#define SPD  ((size_t)SP * DIMN)

typedef __bf16 bf16;
typedef __attribute__((ext_vector_type(16))) __bf16 bf16x16;
typedef __attribute__((ext_vector_type(8)))  float  f32x8;

// ---------------------------------------------------------------- WMMA helper
__device__ __forceinline__ f32x8 wmma_bf16(bf16x16 a, bf16x16 b, f32x8 c) {
    // D = A(16x32) * B(32x16) + C(16x16 f32)
    return __builtin_amdgcn_wmma_f32_16x16x32_bf16(
        /*neg_a=*/false, a, /*neg_b=*/false, b,
        /*c_mod=*/(short)0, c, /*reuse_a=*/false, /*reuse_b=*/false);
}

// A-fragment 16x32 (row m striped per lane, K contiguous per lane).
// Also used for B-fragments: B stored [N, K] row-major. Global or LDS base.
__device__ __forceinline__ bf16x16 ldfrag(const bf16* base, int ld, int r0, int k0) {
    const int lane = threadIdx.x & 31;
    const bf16* p = base + (size_t)(r0 + (lane & 15)) * ld + k0 + ((lane >> 4) << 4);
    return *(const bf16x16*)p;
}

// ---------------------------------------------------------------- CDNA5 async copy
// GV-mode async global->LDS copy, 16 bytes per lane. Tracked by ASYNCcnt.
__device__ __forceinline__ void async_copy16(unsigned lds_off, const void* gaddr) {
    asm volatile("global_load_async_to_lds_b128 %0, %1, off"
                 :: "v"(lds_off), "v"(gaddr) : "memory");
}
#define WAIT_ASYNC(n) asm volatile("s_wait_asynccnt " #n ::: "memory")

typedef __attribute__((address_space(3))) const char* lds_cptr_t;
__device__ __forceinline__ unsigned lds_off_of(const void* p) {
    return (unsigned)(unsigned long long)(lds_cptr_t)(const char*)p;
}

// ---------------------------------------------------------------- CDNA5 TDM
#if __has_builtin(__builtin_amdgcn_tensor_load_to_lds) && \
    __has_builtin(__builtin_amdgcn_s_wait_tensorcnt)
#define HAVE_TDM 1
typedef __attribute__((ext_vector_type(4))) unsigned u32x4;
typedef __attribute__((ext_vector_type(8))) int i32x8;
typedef __attribute__((ext_vector_type(4))) int i32x4;

// 2D tile load: tile_d0 x tile_d1 elements (2B each) from a row-major tensor
// with row stride stride_d0 (elements) into contiguous LDS at lds_byte.
__device__ __forceinline__ void tdm_load_2d(unsigned lds_byte, const void* gptr,
                                            unsigned tile_d0, unsigned tile_d1,
                                            unsigned tensor_d0, unsigned tensor_d1,
                                            unsigned stride_d0) {
    const unsigned long long ga = (unsigned long long)gptr;
    u32x4 g0;
    g0[0] = 1u;                                        // count=1, user descriptor
    g0[1] = lds_byte;                                  // lds_addr
    g0[2] = (unsigned)(ga & 0xffffffffu);              // global_addr[31:0]
    g0[3] = (unsigned)((ga >> 32) & 0x01ffffffu) | (2u << 30);  // addr[56:32], type=2
    i32x8 g1;
    g1[0] = (int)(1u << 16);                           // data_size=1 (2 bytes)
    g1[1] = (int)((tensor_d0 & 0xffffu) << 16);        // tensor_dim0[15:0]
    g1[2] = (int)(((tensor_d0 >> 16) & 0xffffu) | ((tensor_d1 & 0xffffu) << 16));
    g1[3] = (int)(((tensor_d1 >> 16) & 0xffffu) | ((tile_d0 & 0xffffu) << 16));
    g1[4] = (int)(tile_d1 & 0xffffu);                  // tile_dim1 (tile_dim2=0)
    g1[5] = (int)stride_d0;                            // tensor_dim0_stride[31:0]
    g1[6] = 0;
    g1[7] = 0;
    i32x4 zz = {0, 0, 0, 0};
#if __clang_major__ >= 23
    i32x8 z8 = {0, 0, 0, 0, 0, 0, 0, 0};
    __builtin_amdgcn_tensor_load_to_lds(g0, g1, zz, zz, z8, 0);
#else
    __builtin_amdgcn_tensor_load_to_lds(g0, g1, zz, zz, 0);
#endif
}
#endif

// ---------------------------------------------------------------- generic GEMM
// C[M,N] = A[M,K](bf16) * B[N,K]^T(bf16) + bias[N], f32 out.
// grid = (N/128, Mpad/128), block = 256 (8 waves; wave computes 32x64).
// Double-buffered async-to-LDS staging: A tile 128x64, B tile 128x64 per stage.
__global__ __launch_bounds__(256)
void gemm_bf16_wmma(const bf16* __restrict__ A, int lda,
                    const bf16* __restrict__ B, int ldb,
                    const float* __restrict__ bias,
                    float* __restrict__ C, int ldc,
                    int Mstore, int K) {
    __shared__ bf16 As[2][128 * 64] __attribute__((aligned(16)));
    __shared__ bf16 Bs[2][128 * 64] __attribute__((aligned(16)));

    const int tid  = threadIdx.x;
    const int wave = tid >> 5;
    const int lane = tid & 31;
    const int wm = wave & 3, wn = wave >> 2;
    const int m0blk = blockIdx.y * 128;
    const int n0blk = blockIdx.x * 128;

    // per-thread async copy slots: 1024 chunks of 16B per tile, 4 per thread
    unsigned aLds[4], bLds[4];
    const bf16* aP[4];
    const bf16* bP[4];
#pragma unroll
    for (int i = 0; i < 4; ++i) {
        const int c = tid + 256 * i;           // 128 rows x 8 chunks
        const int row = c >> 3, ck = c & 7;
        aLds[i] = (unsigned)(row * 128 + ck * 16);
        bLds[i] = aLds[i];
        aP[i] = A + (size_t)(m0blk + row) * lda + ck * 8;
        bP[i] = B + (size_t)(n0blk + row) * ldb + ck * 8;
    }
    const unsigned asB[2] = { lds_off_of(&As[0][0]), lds_off_of(&As[1][0]) };
    const unsigned bsB[2] = { lds_off_of(&Bs[0][0]), lds_off_of(&Bs[1][0]) };

    // prologue: stage k-step 0 into buffer 0
#pragma unroll
    for (int i = 0; i < 4; ++i) async_copy16(asB[0] + aLds[i], aP[i]);
#pragma unroll
    for (int i = 0; i < 4; ++i) async_copy16(bsB[0] + bLds[i], bP[i]);

    f32x8 acc[2][4] = {};
    const int nsteps = K >> 6;
    int p = 0;
    for (int kt = 0; kt < nsteps; ++kt) {
        if (kt + 1 < nsteps) {
#pragma unroll
            for (int i = 0; i < 4; ++i) {
                aP[i] += 64;
                async_copy16(asB[p ^ 1] + aLds[i], aP[i]);
            }
#pragma unroll
            for (int i = 0; i < 4; ++i) {
                bP[i] += 64;
                async_copy16(bsB[p ^ 1] + bLds[i], bP[i]);
            }
            WAIT_ASYNC(0x8);          // current tile's 8 copies complete
        } else {
            WAIT_ASYNC(0x0);
        }
        __syncthreads();

#pragma unroll
        for (int kh = 0; kh < 2; ++kh) {
            bf16x16 a0 = ldfrag(&As[p][0], 64, wm * 32,      kh * 32);
            bf16x16 a1 = ldfrag(&As[p][0], 64, wm * 32 + 16, kh * 32);
#pragma unroll
            for (int ni = 0; ni < 4; ++ni) {
                bf16x16 bf = ldfrag(&Bs[p][0], 64, wn * 64 + ni * 16, kh * 32);
                acc[0][ni] = wmma_bf16(a0, bf, acc[0][ni]);
                acc[1][ni] = wmma_bf16(a1, bf, acc[1][ni]);
            }
        }

        __syncthreads();              // reads done before buffer reuse
        p ^= 1;
    }

    const int half = lane >> 4, nn = lane & 15;
#pragma unroll
    for (int mi = 0; mi < 2; ++mi)
#pragma unroll
        for (int ni = 0; ni < 4; ++ni)
#pragma unroll
            for (int r = 0; r < 8; ++r) {
                int row = m0blk + wm * 32 + mi * 16 + r + half * 8;
                int col = n0blk + wn * 64 + ni * 16 + nn;
                if (row < Mstore) {
                    float v = acc[mi][ni][r] + (bias ? bias[col] : 0.0f);
                    C[(size_t)row * ldc + col] = v;
                }
            }
}

// ---------------------------------------------------------------- RMS + rope (+z)
__global__ __launch_bounds__(256)
void rms_rope_kernel(const float* __restrict__ Yin, const float* __restrict__ g,
                     const float* __restrict__ freqs,
                     const float* __restrict__ g_kmean,
                     bf16* __restrict__ Rout, float* __restrict__ Z) {
    const int s = blockIdx.x;
    const int tid = threadIdx.x;
    __shared__ float red[256];
    __shared__ float shz[NHN];

    if (s >= S_LEN) {
        for (int i = tid; i < DIMN; i += 256) Rout[(size_t)s * DIMN + i] = (bf16)0.0f;
        if (Z && tid < NHN) Z[s * NHN + tid] = 0.0f;
        return;
    }
    const float* xrow = Yin + (size_t)s * DIMN;

    float ss = 0.0f;
    for (int i = tid; i < DIMN; i += 256) { float v = xrow[i]; ss += v * v; }
    red[tid] = ss;
    if (tid < NHN) shz[tid] = 0.0f;
    for (int off = 128; off > 0; off >>= 1) {
        __syncthreads();
        if (tid < off) red[tid] += red[tid + off];
    }
    __syncthreads();
    const float rms = rsqrtf(red[0] * (1.0f / DIMN) + 1e-6f);

    const int fi = s / (30 * 52);
    const int rem = s % (30 * 52);
    const int hi = rem / 52;
    const int wi = rem % 52;

    for (int p = tid; p < DIMN / 2; p += 256) {
        const int n = p >> 6;          // head
        const int c = p & 63;          // rotary pair index
        const int row = (c < 22) ? fi : (c < 43) ? hi : wi;
        const float cr = freqs[row * 128 + c * 2 + 0];
        const float ci = freqs[row * 128 + c * 2 + 1];
        const int d0 = n * HDN + c * 2;
        const float xr = xrow[d0]     * rms * g[d0];
        const float xi = xrow[d0 + 1] * rms * g[d0 + 1];
        Rout[(size_t)s * DIMN + d0]     = (bf16)(xr * cr - xi * ci);
        Rout[(size_t)s * DIMN + d0 + 1] = (bf16)(xr * ci + xi * cr);
        if (Z) {
            float a = fmaxf(xr, 0.0f) * g_kmean[n * HDN + c * 2]
                    + fmaxf(xi, 0.0f) * g_kmean[n * HDN + c * 2 + 1];
            atomicAdd(&shz[n], a);   // ds_add_f32
        }
    }
    if (Z) {
        __syncthreads();
        if (tid < NHN) Z[s * NHN + tid] = 1.0f / (shz[tid] + 1e-6f);
    }
}

// ---------------------------------------------------------------- flash attention
// grid = (NH, SP/64), block = 128 (4 waves, each wave owns 16 query rows).
// K (32x128) and V^T (128x32) tiles double-buffered in LDS, staged by the
// Tensor Data Mover (one issuing wave, TENSORcnt) when available, else by
// per-lane async copies (ASYNCcnt).
__global__ __launch_bounds__(128)
void flash_attn_kernel(const bf16* __restrict__ rq, const bf16* __restrict__ rk,
                       const bf16* __restrict__ vT, float* __restrict__ xl) {
    __shared__ bf16 Ks[2][32 * 128] __attribute__((aligned(16)));  // [t][d]
    __shared__ bf16 Vs[2][128 * 32] __attribute__((aligned(16)));  // [d][t]
    __shared__ bf16 pbuf[4][16 * 32] __attribute__((aligned(32)));

    const int h = blockIdx.x;
    const int tid = threadIdx.x;
    const int wave = tid >> 5;
    const int lane = tid & 31;
    const int q0 = blockIdx.y * 64 + wave * 16;
    const int half = lane >> 4, nn = lane & 15;
    const float scale = 0.08838834764831845f;   // 128^-0.5

    bf16x16 qf[4];
#pragma unroll
    for (int kk = 0; kk < 4; ++kk) qf[kk] = ldfrag(rq + h * HDN, DIMN, q0, kk * 32);

    const unsigned ksB[2] = { lds_off_of(&Ks[0][0]), lds_off_of(&Ks[1][0]) };
    const unsigned vsB[2] = { lds_off_of(&Vs[0][0]), lds_off_of(&Vs[1][0]) };

#ifdef HAVE_TDM
    const bf16* kBase = rk + h * HDN;                 // row j: kBase + j*DIM
    const bf16* vBase = vT + (size_t)h * HDN * SP;    // row d: vBase + d*SP + j
    if (wave == 0) {
        tdm_load_2d(ksB[0], kBase, 128, 32, DIMN, SP, DIMN);
        tdm_load_2d(vsB[0], vBase, 32, 128, SP, DIMN, SP);
    }
#else
    unsigned kLds[4], vLds[4];
    const bf16* kP[4];
    const bf16* vP[4];
#pragma unroll
    for (int i = 0; i < 4; ++i) {
        const int c = tid + 128 * i;
        const int rowk = c >> 4, colk = c & 15;        // 32 rows x 16 chunks
        kLds[i] = (unsigned)(rowk * 256 + colk * 16);
        kP[i] = rk + (size_t)rowk * DIMN + h * HDN + colk * 8;
        const int rowv = c >> 2, colv = c & 3;         // 128 rows x 4 chunks
        vLds[i] = (unsigned)(rowv * 64 + colv * 16);
        vP[i] = vT + (size_t)(h * HDN + rowv) * SP + colv * 8;
    }
#pragma unroll
    for (int i = 0; i < 4; ++i) async_copy16(ksB[0] + kLds[i], kP[i]);
#pragma unroll
    for (int i = 0; i < 4; ++i) async_copy16(vsB[0] + vLds[i], vP[i]);
#endif

    f32x8 o[8] = {};
    float m_i[8], l_i[8];
#pragma unroll
    for (int r = 0; r < 8; ++r) { m_i[r] = -1e30f; l_i[r] = 0.0f; }

    int p = 0;
    for (int j0 = 0; j0 < S_LEN; j0 += 32) {
        const bool hasNext = (j0 + 32) < S_LEN;
#ifdef HAVE_TDM
        if (wave == 0) {
            if (hasNext) {
                tdm_load_2d(ksB[p ^ 1], kBase + (size_t)(j0 + 32) * DIMN,
                            128, 32, DIMN, SP, DIMN);
                tdm_load_2d(vsB[p ^ 1], vBase + (j0 + 32),
                            32, 128, SP, DIMN, SP);
                __builtin_amdgcn_s_wait_tensorcnt((short)2);
            } else {
                __builtin_amdgcn_s_wait_tensorcnt((short)0);
            }
        }
#else
        if (hasNext) {
#pragma unroll
            for (int i = 0; i < 4; ++i) {
                kP[i] += (size_t)32 * DIMN;
                async_copy16(ksB[p ^ 1] + kLds[i], kP[i]);
            }
#pragma unroll
            for (int i = 0; i < 4; ++i) {
                vP[i] += 32;
                async_copy16(vsB[p ^ 1] + vLds[i], vP[i]);
            }
            WAIT_ASYNC(0x8);
        } else {
            WAIT_ASYNC(0x0);
        }
#endif
        __syncthreads();

        // ---- S = Q K^T (16x32 per wave, K dim = 128)
        f32x8 s0 = {}, s1 = {};
#pragma unroll
        for (int kk = 0; kk < 4; ++kk) {
            bf16x16 b0 = ldfrag(&Ks[p][0], 128, 0,  kk * 32);
            bf16x16 b1 = ldfrag(&Ks[p][0], 128, 16, kk * 32);
            s0 = wmma_bf16(qf[kk], b0, s0);
            s1 = wmma_bf16(qf[kk], b1, s1);
        }
        const int c0 = j0 + nn, c1 = j0 + 16 + nn;

        // ---- online softmax (row = VGPR index + lane-half, cols across 16 lanes)
        float p0[8], p1[8], f[8];
#pragma unroll
        for (int r = 0; r < 8; ++r) {
            float v0 = (c0 < S_LEN) ? s0[r] * scale : -1e30f;
            float v1 = (c1 < S_LEN) ? s1[r] * scale : -1e30f;
            float mx = fmaxf(v0, v1);
            mx = fmaxf(mx, __shfl_xor(mx, 1));
            mx = fmaxf(mx, __shfl_xor(mx, 2));
            mx = fmaxf(mx, __shfl_xor(mx, 4));
            mx = fmaxf(mx, __shfl_xor(mx, 8));
            float mn = fmaxf(m_i[r], mx);
            f[r] = __expf(m_i[r] - mn);
            m_i[r] = mn;
            p0[r] = __expf(v0 - mn);
            p1[r] = __expf(v1 - mn);
            float rs = p0[r] + p1[r];
            rs += __shfl_xor(rs, 1);
            rs += __shfl_xor(rs, 2);
            rs += __shfl_xor(rs, 4);
            rs += __shfl_xor(rs, 8);
            l_i[r] = l_i[r] * f[r] + rs;
        }
#pragma unroll
        for (int nt = 0; nt < 8; ++nt)
#pragma unroll
            for (int r = 0; r < 8; ++r) o[nt][r] *= f[r];

        // ---- transpose P C-layout -> A-layout through per-wave LDS (in-order DS)
#pragma unroll
        for (int r = 0; r < 8; ++r) {
            const int row = r + half * 8;
            pbuf[wave][row * 32 + nn]      = (bf16)p0[r];
            pbuf[wave][row * 32 + 16 + nn] = (bf16)p1[r];
        }
        bf16x16 pa = *(const bf16x16*)&pbuf[wave][nn * 32 + half * 16];

        // ---- O += P V from staged V^T tile
#pragma unroll
        for (int nt = 0; nt < 8; ++nt) {
            bf16x16 vb = *(const bf16x16*)&Vs[p][(nt * 16 + nn) * 32 + half * 16];
            o[nt] = wmma_bf16(pa, vb, o[nt]);
        }
        __syncthreads();              // tile reads done before buffer reuse
        p ^= 1;
    }

    // ---- epilogue: O / l -> xl [SP, DIM]
#pragma unroll
    for (int r = 0; r < 8; ++r) {
        const float inv = 1.0f / l_i[r];
        const int row = q0 + r + half * 8;
#pragma unroll
        for (int nt = 0; nt < 8; ++nt)
            xl[(size_t)row * DIMN + h * HDN + nt * 16 + nn] = o[nt][r] * inv;
    }
}

// ---------------------------------------------------------------- elementwise
__global__ void cvt_pad_kernel(const float* __restrict__ src, bf16* __restrict__ dst,
                               size_t nvalid, size_t ntotal) {
    size_t i = (size_t)blockIdx.x * blockDim.x + threadIdx.x;
    if (i < ntotal) dst[i] = (i < nvalid) ? (bf16)src[i] : (bf16)0.0f;
}

__global__ void gkv_transpose_kernel(const float* __restrict__ gkv, bf16* __restrict__ out) {
    size_t i = (size_t)blockIdx.x * blockDim.x + threadIdx.x;
    const size_t total = (size_t)NHN * HDN * HDN;
    if (i >= total) return;
    const int h = i / (HDN * HDN);
    const int r = i % (HDN * HDN);
    const int m = r / HDN, d = r % HDN;
    out[i] = (bf16)gkv[(size_t)h * HDN * HDN + (size_t)d * HDN + m];
}

__global__ void transpose_v_kernel(const float* __restrict__ v, bf16* __restrict__ vT) {
    size_t i = (size_t)blockIdx.x * blockDim.x + threadIdx.x;
    if (i >= SPD) return;
    const size_t row = i / SP;      // h*HD + d
    const size_t s = i % SP;
    vT[i] = (s < S_LEN) ? (bf16)v[s * DIMN + row] : (bf16)0.0f;
}

__global__ void scale_z_cvt_kernel(const float* __restrict__ t1, const float* __restrict__ z,
                                   bf16* __restrict__ out) {
    size_t i = (size_t)blockIdx.x * blockDim.x + threadIdx.x;
    if (i >= SPD) return;
    const size_t s = i / DIMN;
    const int n = (int)((i % DIMN) >> 7);
    out[i] = (s < S_LEN) ? (bf16)(t1[i] * z[s * NHN + n]) : (bf16)0.0f;
}

__global__ void combine_cvt_kernel(const float* __restrict__ a, const float* __restrict__ b,
                                   bf16* __restrict__ out) {
    size_t i = (size_t)blockIdx.x * blockDim.x + threadIdx.x;
    if (i >= SPD) return;
    const size_t s = i / DIMN;
    out[i] = (s < S_LEN) ? (bf16)(a[i] + b[i]) : (bf16)0.0f;
}

// ---------------------------------------------------------------- host launcher
static inline size_t cdiv_sz(size_t a, size_t b) { return (a + b - 1) / b; }

extern "C" void kernel_launch(void* const* d_in, const int* in_sizes, int n_in,
                              void* d_out, int out_size, void* d_ws, size_t ws_size,
                              hipStream_t stream) {
    const float* x       = (const float*)d_in[0];
    const float* freqs   = (const float*)d_in[1];
    const float* g_kmean = (const float*)d_in[2];
    const float* g_kv    = (const float*)d_in[3];
    const float* Wq = (const float*)d_in[4];  const float* bq = (const float*)d_in[5];
    const float* Wk = (const float*)d_in[6];  const float* bk = (const float*)d_in[7];
    const float* Wv = (const float*)d_in[8];  const float* bv = (const float*)d_in[9];
    const float* Wo = (const float*)d_in[10]; const float* bo = (const float*)d_in[11];
    const float* gq = (const float*)d_in[12]; const float* gk = (const float*)d_in[13];
    const float* Wlin = (const float*)d_in[14]; const float* blin = (const float*)d_in[15];
    float* out = (float*)d_out;

    // ---- workspace carve (256B aligned)
    char* w = (char*)d_ws;
    auto carve = [&](size_t bytes) -> void* {
        void* p = (void*)w;
        w += (bytes + 255) & ~(size_t)255;
        return p;
    };
    bf16* xb    = (bf16*)carve(SPD * 2);                       // x bf16; later reused as t1b
    bf16* wqb   = (bf16*)carve((size_t)DIMN * DIMN * 2);
    bf16* wkb   = (bf16*)carve((size_t)DIMN * DIMN * 2);
    bf16* wvb   = (bf16*)carve((size_t)DIMN * DIMN * 2);
    bf16* wob   = (bf16*)carve((size_t)DIMN * DIMN * 2);
    bf16* wlinb = (bf16*)carve((size_t)HDN * HDN * 2);
    bf16* gkvTb = (bf16*)carve((size_t)NHN * HDN * HDN * 2);
    float* bufA = (float*)carve(SPD * 4);                      // q f32 -> t1
    float* bufB = (float*)carve(SPD * 4);                      // k f32 -> xg2
    float* bufC = (float*)carve(SPD * 4);                      // v f32 -> x_local
    bf16* rqb   = (bf16*)carve(SPD * 2);
    bf16* rkb   = (bf16*)carve(SPD * 2);                       // later reused as ycomb
    bf16* vTb   = (bf16*)carve(SPD * 2);
    float* zb   = (float*)carve((size_t)SP * NHN * 4);
    (void)ws_size; (void)n_in; (void)in_sizes; (void)out_size;

    const size_t EW = 256;
    const size_t W2 = (size_t)DIMN * DIMN;

    // ---- f32 -> bf16 conversions
    cvt_pad_kernel<<<cdiv_sz(SPD, EW), EW, 0, stream>>>(x, xb, (size_t)S_LEN * DIMN, SPD);
    cvt_pad_kernel<<<cdiv_sz(W2, EW), EW, 0, stream>>>(Wq, wqb, W2, W2);
    cvt_pad_kernel<<<cdiv_sz(W2, EW), EW, 0, stream>>>(Wk, wkb, W2, W2);
    cvt_pad_kernel<<<cdiv_sz(W2, EW), EW, 0, stream>>>(Wv, wvb, W2, W2);
    cvt_pad_kernel<<<cdiv_sz(W2, EW), EW, 0, stream>>>(Wo, wob, W2, W2);
    cvt_pad_kernel<<<cdiv_sz((size_t)HDN * HDN, EW), EW, 0, stream>>>(
        Wlin, wlinb, (size_t)HDN * HDN, (size_t)HDN * HDN);
    gkv_transpose_kernel<<<cdiv_sz((size_t)NHN * HDN * HDN, EW), EW, 0, stream>>>(g_kv, gkvTb);

    // ---- QKV projections (WMMA GEMM, M=SP, N=DIM, K=DIM)
    const dim3 gBig(DIMN / 128, SP / 128);
    gemm_bf16_wmma<<<gBig, 256, 0, stream>>>(xb, DIMN, wqb, DIMN, bq, bufA, DIMN, SP, DIMN);
    gemm_bf16_wmma<<<gBig, 256, 0, stream>>>(xb, DIMN, wkb, DIMN, bk, bufB, DIMN, SP, DIMN);
    gemm_bf16_wmma<<<gBig, 256, 0, stream>>>(xb, DIMN, wvb, DIMN, bv, bufC, DIMN, SP, DIMN);

    // ---- RMS + rope (+ z for Q), V transpose to [h*HD+d][s]
    rms_rope_kernel<<<SP, 256, 0, stream>>>(bufA, gq, freqs, g_kmean, rqb, zb);
    rms_rope_kernel<<<SP, 256, 0, stream>>>(bufB, gk, freqs, g_kmean, rkb, nullptr);
    transpose_v_kernel<<<cdiv_sz(SPD, EW), EW, 0, stream>>>(bufC, vTb);

    // ---- flash attention -> x_local (bufC)
    flash_attn_kernel<<<dim3(NHN, SP / 64), 128, 0, stream>>>(rqb, rkb, vTb, bufC);

    // ---- linear branch: t1 = rq @ g_kv (per head)  (M=SP, N=128, K=128)
    const dim3 gHead(1, SP / 128);
    for (int h = 0; h < NHN; ++h)
        gemm_bf16_wmma<<<gHead, 256, 0, stream>>>(
            rqb + h * HDN, DIMN, gkvTb + (size_t)h * HDN * HDN, HDN,
            nullptr, bufA + h * HDN, DIMN, SP, HDN);

    // t1b = bf16(t1 * z)   (reuses xb)
    scale_z_cvt_kernel<<<cdiv_sz(SPD, EW), EW, 0, stream>>>(bufA, zb, xb);

    // xg = t1b @ Wlin^T + blin (per head) -> bufB
    for (int h = 0; h < NHN; ++h)
        gemm_bf16_wmma<<<gHead, 256, 0, stream>>>(
            xb + h * HDN, DIMN, wlinb, HDN, blin, bufB + h * HDN, DIMN, SP, HDN);

    // ---- ycomb = bf16(x_local + xg)  (reuses rkb)
    combine_cvt_kernel<<<cdiv_sz(SPD, EW), EW, 0, stream>>>(bufC, bufB, rkb);

    // ---- out = ycomb @ Wo^T + bo   (store guarded to S rows)
    gemm_bf16_wmma<<<gBig, 256, 0, stream>>>(rkb, DIMN, wob, DIMN, bo, out, DIMN, S_LEN, DIMN);
}
